// CarbonGNN_10290741641494
// MI455X (gfx1250) — matrix operand order repaired and checked by reading, compiled, not run on hardware
//
#include <hip/hip_runtime.h>
#include <hip/hip_bf16.h>

// ---------------------------------------------------------------------------
// CarbonGNN on gfx1250: 3x SAGEConv (mean aggr) + edge MLP + classifier.
// GEMMs: v_wmma_f32_16x16x32_bf16 (wave32, fp32 accumulate).
// Weights pre-packed in global memory into the exact WMMA B-fragment layout
// (u32 = bf16 pair (K,K+1), index [kchunk][col][pair]) -> B frag = 2x b128.
// A tiles staged in LDS with 16B-aligned rows -> A frag = 2x ds_load_b128.
// All B fragments for a K-step are issued as one load burst before the WMMA
// chain so s_wait_loadcnt drains progressively while earlier WMMAs execute.
// ---------------------------------------------------------------------------

typedef __attribute__((ext_vector_type(16))) __bf16 v16bf;
typedef __attribute__((ext_vector_type(8)))  float  v8f;

union frag_cast { v16bf v; uint4 q[2]; };
union bf2pack  { __bf16 b[2]; unsigned int u; };

// ---- WMMA fragment loaders (CDNA5 ISA 7.12.2 layouts) ---------------------
// A: 16x32 (MxK) bf16 tile in LDS, row stride 40 elements (80B, 16B aligned).
// lane m=lane&15 -> row M; lanes 0-15: K{0..7,16..23}, lanes 16-31: +8.
__device__ inline v16bf frag_a_pk(const __bf16* tile /* row 0 of 16-row tile */) {
  int lane = threadIdx.x & 31;
  int m  = lane & 15;
  int kh = (lane >> 4) * 8;
  const uint4* q = (const uint4*)(tile + m * 40 + kh);
  frag_cast r;
  r.q[0] = q[0];   // K pairs kbase 0,2,4,6   (+kh)
  r.q[1] = q[2];   // K pairs kbase 16,18,20,22 (+kh)
  return r.v;
}

// B: 32x16 (KxN) from packed global weights.
// base = Wpk + (kc*Nc + ct*16)*16 (u32 units). u[v] = pair(K=2v+8*kh (v<4), 16+2(v-4)+8*kh (v>=4), col n).
__device__ inline v16bf frag_b_pk(const unsigned int* base) {
  int lane = threadIdx.x & 31;
  int n  = lane & 15;
  int kh = lane >> 4;
  const uint4* q = (const uint4*)(base + n * 16 + kh * 4);
  frag_cast r;
  r.q[0] = q[0];
  r.q[1] = q[2];
  return r.v;
}

__device__ inline v8f wmma_bf16(v16bf a, v16bf b, v8f c) {
  return __builtin_amdgcn_wmma_f32_16x16x32_bf16(
      false, a, false, b, (short)0, c, false, false);
}

// pack 16 floats (optionally scaled) into 8 u32 bf16-pairs, store as 2x uint4
__device__ inline void pack_row16(const float* s, float scale, __bf16* dst) {
  unsigned int u[8];
#pragma unroll
  for (int j = 0; j < 8; ++j) {
    bf2pack pk;
    pk.b[0] = (__bf16)(s[2 * j] * scale);
    pk.b[1] = (__bf16)(s[2 * j + 1] * scale);
    u[j] = pk.u;
  }
  uint4* d = (uint4*)dst;
  d[0] = make_uint4(u[0], u[1], u[2], u[3]);
  d[1] = make_uint4(u[4], u[5], u[6], u[7]);
}

// ---- small utility kernels -------------------------------------------------
__global__ void zero_f32(float* __restrict__ p, long n) {
  long i = (long)blockIdx.x * blockDim.x + threadIdx.x;
  if (i < n) p[i] = 0.0f;
}

__global__ void count_edges(const int* __restrict__ dst, float* __restrict__ cnt, int E) {
  int e = blockIdx.x * blockDim.x + threadIdx.x;
  if (e < E) atomicAdd(&cnt[dst[e]], 1.0f);
}

__global__ void finalize_inv(float* __restrict__ cnt, int N) {
  int i = blockIdx.x * blockDim.x + threadIdx.x;
  if (i < N) cnt[i] = 1.0f / fmaxf(cnt[i], 1.0f);
}

// agg[dst] += h[src], F = 128, 4 floats per thread
__global__ void scatter_add128(const float* __restrict__ h, const int* __restrict__ src,
                               const int* __restrict__ dst, float* __restrict__ agg, int E) {
  long t = (long)blockIdx.x * blockDim.x + threadIdx.x;
  long total = (long)E * 32;
  if (t >= total) return;
  int e = (int)(t >> 5);
  int q = (int)(t & 31) * 4;
  const float* s = h + (size_t)src[e] * 128 + q;
  float* d = agg + (size_t)dst[e] * 128 + q;
  atomicAdd(d + 0, s[0]);
  atomicAdd(d + 1, s[1]);
  atomicAdd(d + 2, s[2]);
  atomicAdd(d + 3, s[3]);
}

// Pack logical weight matrix (rows x Nc, row k<split from A else B[k-split])
// into WMMA B-fragment layout: out[kc*(Nc*16) + n*16 + p] = pair(K=kc*32+2p, col n).
__global__ void pack_pairs(const float* __restrict__ A, const float* __restrict__ B,
                           unsigned int* __restrict__ out, int split, int rows, int Nc) {
  int i = blockIdx.x * blockDim.x + threadIdx.x;
  int total = (rows >> 1) * Nc;           // one u32 pair per thread
  if (i >= total) return;
  int kc  = i / (Nc * 16);
  int rem = i - kc * (Nc * 16);
  int n = rem >> 4;
  int p = rem & 15;
  int k = kc * 32 + 2 * p;                // even; (k, k+1) on same side of split
  float v0 = (k     < split) ? A[k * Nc + n]           : B[(k - split) * Nc + n];
  float v1 = (k + 1 < split) ? A[(k + 1) * Nc + n]     : B[(k + 1 - split) * Nc + n];
  bf2pack pk;
  pk.b[0] = (__bf16)v0;
  pk.b[1] = (__bf16)v1;
  out[i] = pk.u;
}

// ---- fused SAGE layer GEMM: out = relu?([mean|h] @ Wcat + b) --------------
// K = 256 (mean 0..127 | root 128..255). Block: 256 thr = 8 waves, 128 rows.
template <int NCOLS, int RELU>
__global__ __launch_bounds__(256) void sage_gemm(
    const float* __restrict__ agg, const float* __restrict__ inv,
    const float* __restrict__ hin, const unsigned int* __restrict__ Wpk,
    const float* __restrict__ bias, float* __restrict__ out, int N) {
  __shared__ __align__(16) __bf16 As[128][40];  // 32-K chunk, 16B-aligned rows

  const int tid  = threadIdx.x;
  const int wave = tid >> 5;
  const int lane = tid & 31;
  const int row_base = blockIdx.x * 128;
  constexpr int NT = NCOLS / 16;

  v8f acc[NT] = {};

#pragma unroll 1
  for (int kc = 0; kc < 8; ++kc) {
    const int k0 = kc * 32;
    // stage A chunk: 128 rows x 32 feats; 16 contiguous feats per thread
    {
      int r  = tid >> 1;
      int c0 = (tid & 1) * 16;
      int node = row_base + r;
      if (node >= N) node = N - 1;              // clamp (writes guarded later)
      if (k0 < 128) {                           // mean part
        pack_row16(agg + (size_t)node * 128 + k0 + c0, inv[node], &As[r][c0]);
      } else {                                  // root part
        pack_row16(hin + (size_t)node * 128 + (k0 - 128) + c0, 1.0f, &As[r][c0]);
      }
    }
    __syncthreads();

    // burst-issue all B fragments, then A, then drain into the WMMA chain
    v16bf bfr[NT];
#pragma unroll
    for (int ct = 0; ct < NT; ++ct)
      bfr[ct] = frag_b_pk(Wpk + ((size_t)kc * NCOLS + ct * 16) * 16);
    v16bf a = frag_a_pk(&As[wave * 16][0]);
#pragma unroll
    for (int ct = 0; ct < NT; ++ct)
      acc[ct] = wmma_bf16(a, bfr[ct], acc[ct]);
    __syncthreads();
  }

  // writeback: VGPR r holds (M = r + 8*(lane>>4), N = lane&15)
  const int half = lane >> 4;
  const int n = lane & 15;
#pragma unroll
  for (int ct = 0; ct < NT; ++ct) {
    int col = ct * 16 + n;
    float b = bias[col];
#pragma unroll
    for (int r = 0; r < 8; ++r) {
      int node = row_base + wave * 16 + half * 8 + r;
      if (node < N) {
        float v = acc[ct][r] + b;
        if (RELU) v = fmaxf(v, 0.0f);
        out[(size_t)node * NCOLS + col] = v;
      }
    }
  }
}

// ---- fused edge head: flow = relu([ne[src]|ne[dst]] @ Wf1 + b1) @ Wf2 + b2 -
__global__ __launch_bounds__(256) void edge_mlp(
    const float* __restrict__ ne, const int* __restrict__ src,
    const int* __restrict__ dst, const unsigned int* __restrict__ Wf1p,
    const float* __restrict__ bf1, const float* __restrict__ Wf2,
    const float* __restrict__ bf2, float* __restrict__ flows, int E) {
  __shared__ __align__(16) __bf16 As[128][40];
  __shared__ int   sidx[128], didx[128];
  __shared__ float sW2[128], sb1[128];

  const int tid  = threadIdx.x;
  const int wave = tid >> 5;
  const int lane = tid & 31;
  const int e_base = blockIdx.x * 128;

  if (tid < 128) {
    int e = e_base + tid;
    if (e >= E) e = E - 1;
    sidx[tid] = src[e];
    didx[tid] = dst[e];
    sW2[tid] = Wf2[tid];
    sb1[tid] = bf1[tid];
  }
  __syncthreads();

  // per-thread gather base pointers (hoisted out of the K loop)
  const int rr = tid >> 1;
  const int cc0 = (tid & 1) * 16;
  const float* psrc = ne + (size_t)sidx[rr] * 64;
  const float* pdst = ne + (size_t)didx[rr] * 64;

  v8f acc[8] = {};
#pragma unroll 1
  for (int kc = 0; kc < 4; ++kc) {          // K = 128 = 2*OUT
    const int f0 = kc * 32 + cc0;           // 16-aligned: whole group one side
    {   // stage edge embeddings (gather + cvt); 16 feats per thread
      const float* s = (f0 < 64) ? (psrc + f0) : (pdst + (f0 - 64));
      pack_row16(s, 1.0f, &As[rr][cc0]);
    }
    __syncthreads();

    v16bf bfr[8];
#pragma unroll
    for (int ct = 0; ct < 8; ++ct)
      bfr[ct] = frag_b_pk(Wf1p + ((size_t)kc * 128 + ct * 16) * 16);
    v16bf a = frag_a_pk(&As[wave * 16][0]);
#pragma unroll
    for (int ct = 0; ct < 8; ++ct)
      acc[ct] = wmma_bf16(a, bfr[ct], acc[ct]);
    __syncthreads();
  }

  // bias + relu + dot with Wf2, reduce over the 16 N-lanes per half
  const int half = lane >> 4;
  const int n = lane & 15;
  float partial[8];
#pragma unroll
  for (int r = 0; r < 8; ++r) partial[r] = 0.0f;
#pragma unroll
  for (int ct = 0; ct < 8; ++ct) {
    int col = ct * 16 + n;
    float b = sb1[col], w2 = sW2[col];
#pragma unroll
    for (int r = 0; r < 8; ++r)
      partial[r] += fmaxf(acc[ct][r] + b, 0.0f) * w2;
  }
#pragma unroll
  for (int m = 1; m < 16; m <<= 1)
#pragma unroll
    for (int r = 0; r < 8; ++r)
      partial[r] += __shfl_xor(partial[r], m, 32);

  if (n == 0) {
    float b2 = bf2[0];
#pragma unroll
    for (int r = 0; r < 8; ++r) {
      int e = e_base + wave * 16 + half * 8 + r;
      if (e < E) flows[e] = partial[r] + b2;
    }
  }
}

// ---- classifier: classes = ne @ Wc + bc (tiny, VALU) ----------------------
__global__ void classify(const float* __restrict__ ne, const float* __restrict__ Wc,
                         const float* __restrict__ bc, float* __restrict__ out, int N) {
  __shared__ float sW[256];
  __shared__ float sb[4];
  int tid = threadIdx.x;
  if (tid < 256) sW[tid] = Wc[tid];
  if (tid < 4) sb[tid] = bc[tid];
  __syncthreads();
  int g = blockIdx.x * blockDim.x + tid;
  if (g < N * 4) {
    int node = g >> 2, c = g & 3;
    const float* row = ne + (size_t)node * 64;
    float s = sb[c];
#pragma unroll 8
    for (int k = 0; k < 64; ++k) s += row[k] * sW[k * 4 + c];
    out[g] = s;
  }
}

// ---------------------------------------------------------------------------
extern "C" void kernel_launch(void* const* d_in, const int* in_sizes, int n_in,
                              void* d_out, int out_size, void* d_ws, size_t ws_size,
                              hipStream_t stream) {
  const int N = in_sizes[0] / 128;     // 50000
  const int E = in_sizes[1] / 2;       // 800000

  const float* x    = (const float*)d_in[0];
  const int*   eidx = (const int*)d_in[1];
  const int*   src  = eidx;
  const int*   dst  = eidx + E;
  const float* Wl0 = (const float*)d_in[2];
  const float* Wr0 = (const float*)d_in[3];
  const float* b0  = (const float*)d_in[4];
  const float* Wl1 = (const float*)d_in[5];
  const float* Wr1 = (const float*)d_in[6];
  const float* b1  = (const float*)d_in[7];
  const float* Wl2 = (const float*)d_in[8];
  const float* Wr2 = (const float*)d_in[9];
  const float* b2  = (const float*)d_in[10];
  const float* Wf1 = (const float*)d_in[11];
  const float* bf1 = (const float*)d_in[12];
  const float* Wf2 = (const float*)d_in[13];
  const float* bf2 = (const float*)d_in[14];
  const float* Wc  = (const float*)d_in[15];
  const float* bc  = (const float*)d_in[16];

  // output regions: [ne N*64][flows E][classes N*4]
  float* ne      = (float*)d_out;
  float* flows   = ne + (size_t)N * 64;
  float* classes = flows + (size_t)E;

  // workspace layout
  char* w = (char*)d_ws;
  auto carve = [&](size_t bytes) {
    char* p = w;
    w += (bytes + 255) & ~(size_t)255;
    return p;
  };
  float*        cnt  = (float*)carve((size_t)N * 4);
  float*        agg  = (float*)carve((size_t)N * 128 * 4);
  float*        hA   = (float*)carve((size_t)N * 128 * 4);
  float*        hB   = (float*)carve((size_t)N * 128 * 4);
  unsigned int* Wp0  = (unsigned int*)carve(128 * 128 * 4);  // 256x128 pairs
  unsigned int* Wp1  = (unsigned int*)carve(128 * 128 * 4);
  unsigned int* Wp2  = (unsigned int*)carve(128 * 64 * 4);   // 256x64 pairs
  unsigned int* Wf1p = (unsigned int*)carve(64 * 128 * 4);   // 128x128 pairs

  const int T = 256;
  auto blocks = [](long n, int t) { return (unsigned)((n + t - 1) / t); };

  // degree counts -> inverse
  zero_f32<<<blocks(N, T), T, 0, stream>>>(cnt, N);
  count_edges<<<blocks(E, T), T, 0, stream>>>(dst, cnt, E);
  finalize_inv<<<blocks(N, T), T, 0, stream>>>(cnt, N);

  // weight packing (f32 -> bf16 pairs in WMMA B-fragment layout)
  pack_pairs<<<blocks(128 * 128, T), T, 0, stream>>>(Wl0, Wr0, Wp0, 128, 256, 128);
  pack_pairs<<<blocks(128 * 128, T), T, 0, stream>>>(Wl1, Wr1, Wp1, 128, 256, 128);
  pack_pairs<<<blocks(128 * 64,  T), T, 0, stream>>>(Wl2, Wr2, Wp2, 128, 256, 64);
  pack_pairs<<<blocks(64 * 128,  T), T, 0, stream>>>(Wf1, Wf1, Wf1p, 128, 128, 128);

  const long aggN = (long)N * 128;
  const unsigned gemm_grid = blocks(N, 128);

  // layer 0: x -> hA
  zero_f32<<<blocks(aggN, T), T, 0, stream>>>(agg, aggN);
  scatter_add128<<<blocks((long)E * 32, T), T, 0, stream>>>(x, src, dst, agg, E);
  sage_gemm<128, 1><<<gemm_grid, T, 0, stream>>>(agg, cnt, x, Wp0, b0, hA, N);

  // layer 1: hA -> hB
  zero_f32<<<blocks(aggN, T), T, 0, stream>>>(agg, aggN);
  scatter_add128<<<blocks((long)E * 32, T), T, 0, stream>>>(hA, src, dst, agg, E);
  sage_gemm<128, 1><<<gemm_grid, T, 0, stream>>>(agg, cnt, hA, Wp1, b1, hB, N);

  // layer 2: hB -> ne (no relu)
  zero_f32<<<blocks(aggN, T), T, 0, stream>>>(agg, aggN);
  scatter_add128<<<blocks((long)E * 32, T), T, 0, stream>>>(hB, src, dst, agg, E);
  sage_gemm<64, 0><<<gemm_grid, T, 0, stream>>>(agg, cnt, hB, Wp2, b2, ne, N);

  // edge head + classifier
  edge_mlp<<<blocks(E, 128), T, 0, stream>>>(ne, src, dst, Wf1p, bf1, Wf2, bf2, flows, E);
  classify<<<blocks((long)N * 4, T), T, 0, stream>>>(ne, Wc, bc, classes, N);
}
